// DecodePredictions_13477607374982
// MI455X (gfx1250) — compile-verified, matrix operand order checked by reading, not run on hardware
//
#include <hip/hip_runtime.h>

#define N_ANCH   21824
#define NCLS     80
#define NBATCH   16
#define KCAND    256
#define MAXPC    100
#define MAXDET   100
#define SEG1     86      // ceil(21824/256)
#define SEG2     32      // ceil(8000/256)
#define CONF_THR 0.05f
#define IOU_THR  0.5f

typedef __attribute__((ext_vector_type(16))) __bf16 v16bf;
typedef __attribute__((ext_vector_type(8)))  float  v8f;

// wave32 argmax reduce (larger value wins; tie -> smaller index)
__device__ __forceinline__ void waveArgmax(float& v, int& i) {
#pragma unroll
  for (int off = 16; off > 0; off >>= 1) {
    float ov = __shfl_xor(v, off, 32);
    int   oi = __shfl_xor(i, off, 32);
    if (ov > v || (ov == v && oi < i)) { v = ov; i = oi; }
  }
}

// One block per (class c = blockIdx.x, batch b = blockIdx.y), 256 threads.
__global__ void nms_class_kernel(const float* __restrict__ regs,
                                 const float* __restrict__ ctrs,
                                 const float* __restrict__ clfs,
                                 float* __restrict__ ws_scores,   // [B][C][100]
                                 float* __restrict__ ws_boxes) {  // [B][C][100][4]
  const int c    = blockIdx.x;
  const int b    = blockIdx.y;
  const int tid  = threadIdx.x;
  const int lane = tid & 31;
  const int wave = tid >> 5;

  extern __shared__ char smemRaw[];
  float* sS    = (float*)smemRaw;               // N_ANCH scores
  float* rv    = (float*)(sS + N_ANCH);         // 8 per-wave max vals
  int*   ri    = (int*)(rv + 8);                // 8 per-wave max idx
  float* cSc   = (float*)(ri + 8);              // 256 cand scores (later: kept)
  int*   cIx   = (int*)(cSc + KCAND);           // 256 cand anchor indices
  float* cBox  = (float*)(cIx + KCAND);         // 256*4 cand boxes (y1,x1,y2,x2)
  int*   keepS = (int*)(cBox + KCAND * 4);      // 256 keep flags

  // ---- 1. fused score decode into LDS: score = clf * ctr ----
  const long long bn0 = (long long)b * N_ANCH;
  for (int n = tid; n < N_ANCH; n += 256) {
    float ctr = ctrs[bn0 + n];
    float cf  = clfs[(bn0 + n) * NCLS + c];
    sS[n] = cf * ctr;
  }
  __syncthreads();

  // ---- 2. top-256 (sorted desc, ties by index) via iterative argmax ----
  float lv = -1.0f; int li = 0x7fffffff;
  {
    int s0 = tid * SEG1, s1 = min(s0 + SEG1, N_ANCH);
    for (int n = s0; n < s1; ++n) { float v = sS[n]; if (v > lv) { lv = v; li = n; } }
  }
  for (int it = 0; it < KCAND; ++it) {
    float wv = lv; int wi = li;
    waveArgmax(wv, wi);
    if (lane == 0) { rv[wave] = wv; ri[wave] = wi; }
    __syncthreads();
    float bv = rv[0]; int bi = ri[0];
#pragma unroll
    for (int k = 1; k < 8; ++k) {
      float v = rv[k]; int i2 = ri[k];
      if (v > bv || (v == bv && i2 < bi)) { bv = v; bi = i2; }
    }
    if (tid == 0) { cSc[it] = bv; cIx[it] = bi; sS[bi] = -1.0f; }
    __syncthreads();
    if (bi >= tid * SEG1 && bi < tid * SEG1 + SEG1) {   // winner's owner rescans
      lv = -1.0f; li = 0x7fffffff;
      int s0 = tid * SEG1, s1 = min(s0 + SEG1, N_ANCH);
      for (int n = s0; n < s1; ++n) { float v = sS[n]; if (v > lv) { lv = v; li = n; } }
    }
  }
  __syncthreads();

  // ---- 3. decode candidate boxes (thread j = candidate j) ----
  float y1, x1, y2, x2;
  {
    int n = cIx[tid];
    int base, Hs, st;
    if      (n < 16384) { base = 0;     Hs = 128; st = 8;   }
    else if (n < 20480) { base = 16384; Hs = 64;  st = 16;  }
    else if (n < 21504) { base = 20480; Hs = 32;  st = 32;  }
    else if (n < 21760) { base = 21504; Hs = 16;  st = 64;  }
    else                { base = 21760; Hs = 8;   st = 128; }
    int loc = n - base;
    int q   = loc / Hs;
    float gx = (float)q;
    float gy = (float)(loc - q * Hs);
    const float4 r = ((const float4*)regs)[bn0 + n];
    float fs = (float)st;
    x1 = (gx - r.x) * fs;
    x2 = (gx + r.y) * fs;
    y1 = (gy - r.z) * fs;
    y2 = (gy + r.w) * fs;
    cBox[tid * 4 + 0] = y1; cBox[tid * 4 + 1] = x1;
    cBox[tid * 4 + 2] = y2; cBox[tid * 4 + 3] = x2;
    keepS[tid] = (cSc[tid] > CONF_THR) ? 1 : 0;
  }
  __syncthreads();

  // ---- 4. sequential NMS (reference semantics) ----
  const float areaJ = (y2 - y1) * (x2 - x1);
  for (int i = 0; i < KCAND; ++i) {
    if (keepS[i]) {                       // uniform branch across block
      if (tid > i && keepS[tid]) {
        float iy1 = cBox[i*4+0], ix1 = cBox[i*4+1];
        float iy2 = cBox[i*4+2], ix2 = cBox[i*4+3];
        float areaI = (iy2 - iy1) * (ix2 - ix1);
        float ih = fmaxf(fminf(iy2, y2) - fmaxf(iy1, y1), 0.0f);
        float iw = fmaxf(fminf(ix2, x2) - fmaxf(ix1, x1), 0.0f);
        float inter = ih * iw;
        float uni   = areaI + areaJ - inter;
        if (inter / fmaxf(uni, 1e-8f) > IOU_THR) keepS[tid] = 0;
      }
    }
    __syncthreads();
  }

  // ---- 5. kept scores ----
  cSc[tid] = keepS[tid] ? cSc[tid] : 0.0f;
  __syncthreads();

  // ---- 6. WMMA rank-based top-100: rank[j] = #{i: s_i>s_j or (== and i<j)} ----
  // colsum of 0/1 cmp matrix: D = ones(16x32,bf16) x cmp(32x16,bf16), f32 acc.
  // K-chunk loop kept rolled (#pragma unroll 1) so only ONE fragment is live at
  // a time -> no scratch spills (round-1 asm showed Folded Reloads here).
  const __bf16 one_bf  = (__bf16)1.0f;
  const __bf16 zero_bf = (__bf16)0.0f;
  v16bf ones;
#pragma unroll
  for (int t = 0; t < 16; ++t) ones[t] = one_bf;

#pragma unroll 1
  for (int g = wave * 2; g < wave * 2 + 2; ++g) {     // 2 column groups per wave
    const int col  = g * 16 + (lane & 15);            // B-matrix column (N)
    const int koff = (lane < 16) ? 0 : 16;            // half-wave K split
    const float sj = cSc[col];
    v8f acc = {};
#pragma unroll 1
    for (int ch = 0; ch < 8; ++ch) {                  // 8 * 32 = 256 rows
      v16bf bf;
#pragma unroll
      for (int t = 0; t < 16; ++t) {
        int i = ch * 32 + koff + t;
        float si = cSc[i];
        bool cm = (si > sj) || (si == sj && i < col);
        bf[t] = cm ? one_bf : zero_bf;
      }
      acc = __builtin_amdgcn_wmma_f32_16x16x32_bf16(
          false, ones, false, bf, (short)0, acc, false, false);
    }
    // row M=0 lives in acc[0], lanes 0..15 (N = lane)
    if (lane < 16) {
      int r = (int)(acc[0] + 0.5f);
      if (r < MAXPC) {                                // ranks unique -> slots 0..99 filled
        int sIdx = (b * NCLS + c) * MAXPC + r;
        ws_scores[sIdx]        = cSc[col];
        ws_boxes[sIdx*4 + 0]   = cBox[col*4 + 0];
        ws_boxes[sIdx*4 + 1]   = cBox[col*4 + 1];
        ws_boxes[sIdx*4 + 2]   = cBox[col*4 + 2];
        ws_boxes[sIdx*4 + 3]   = cBox[col*4 + 3];
      }
    }
  }
}

// One block per batch: top-100 of 8000, write final outputs.
__global__ void final_topk_kernel(const float* __restrict__ ws_scores,
                                  const float* __restrict__ ws_boxes,
                                  float* __restrict__ out) {
  const int b = blockIdx.x;
  const int tid = threadIdx.x;
  const int lane = tid & 31, wave = tid >> 5;
  __shared__ float sS[NCLS * MAXPC];
  __shared__ float rv[8];
  __shared__ int   ri[8];
  for (int n = tid; n < NCLS * MAXPC; n += 256) sS[n] = ws_scores[b * NCLS * MAXPC + n];
  __syncthreads();

  float lv = -1.0f; int li = 0x7fffffff;
  {
    int s0 = tid * SEG2, s1 = min(s0 + SEG2, NCLS * MAXPC);
    for (int n = s0; n < s1; ++n) { float v = sS[n]; if (v > lv) { lv = v; li = n; } }
  }
  float* outB = out;                              // [16][100][4]
  float* outL = out + NBATCH * MAXDET * 4;        // [16][100]
  float* outS = outL + NBATCH * MAXDET;           // [16][100]

  for (int it = 0; it < MAXDET; ++it) {
    float wv = lv; int wi = li;
    waveArgmax(wv, wi);
    if (lane == 0) { rv[wave] = wv; ri[wave] = wi; }
    __syncthreads();
    float bv = rv[0]; int bi = ri[0];
#pragma unroll
    for (int k = 1; k < 8; ++k) {
      float v = rv[k]; int i2 = ri[k];
      if (v > bv || (v == bv && i2 < bi)) { bv = v; bi = i2; }
    }
    if (tid == 0) {
      int cls = bi / MAXPC;
      float m = (bv > 0.0f) ? 1.0f : 0.0f;
      int src = (b * NCLS * MAXPC + bi) * 4;
      int dst = b * MAXDET + it;
      outB[dst*4 + 0] = ws_boxes[src + 0] * m;
      outB[dst*4 + 1] = ws_boxes[src + 1] * m;
      outB[dst*4 + 2] = ws_boxes[src + 2] * m;
      outB[dst*4 + 3] = ws_boxes[src + 3] * m;
      outL[dst] = (float)cls;
      outS[dst] = bv;
      sS[bi] = -1.0f;
    }
    __syncthreads();
    if (bi >= tid * SEG2 && bi < tid * SEG2 + SEG2) {
      lv = -1.0f; li = 0x7fffffff;
      int s0 = tid * SEG2, s1 = min(s0 + SEG2, NCLS * MAXPC);
      for (int n = s0; n < s1; ++n) { float v = sS[n]; if (v > lv) { lv = v; li = n; } }
    }
  }
}

extern "C" void kernel_launch(void* const* d_in, const int* in_sizes, int n_in,
                              void* d_out, int out_size, void* d_ws, size_t ws_size,
                              hipStream_t stream) {
  (void)in_sizes; (void)n_in; (void)out_size; (void)ws_size;
  const float* regs = (const float*)d_in[0];   // (B,N,4)
  const float* ctrs = (const float*)d_in[1];   // (B,N,1)
  const float* clfs = (const float*)d_in[2];   // (B,N,80)

  float* wsS = (float*)d_ws;                          // [B][C][100]    scores
  float* wsB = wsS + NBATCH * NCLS * MAXPC;           // [B][C][100][4] boxes

  size_t smem = (size_t)N_ANCH * 4        // scores
              + 8 * 4 + 8 * 4             // rv, ri
              + KCAND * 4 + KCAND * 4     // cand scores, cand idx
              + KCAND * 16                // cand boxes
              + KCAND * 4;                // keep flags   (~94.4 KB; 3 blocks/WGP @320KB)

  nms_class_kernel<<<dim3(NCLS, NBATCH), 256, smem, stream>>>(regs, ctrs, clfs, wsS, wsB);
  final_topk_kernel<<<dim3(NBATCH), 256, 0, stream>>>(wsS, wsB, (float*)d_out);
}
